// GCL_39951785787492
// MI455X (gfx1250) — matrix operand order repaired and checked by reading, not compiled, not run on hardware
//
#include <hip/hip_runtime.h>
#include <hip/hip_bf16.h>

typedef _Float16 v16h __attribute__((ext_vector_type(16)));
typedef float    v8f  __attribute__((ext_vector_type(8)));

#define WV 4            // waves per block (wave32)
#define DD 128          // feature dim

// ---------------------------------------------------------------------------
// Pack fp32 weight W[K][N] (row-major) into f16 WMMA B-fragment tiles.
// Tile (kt,nt) -> 512 halves:
//   dst[tile*512 + lane*16 + j] = W[kt*32 + (j/8)*16 + (lane/16)*8 + j%8][nt*16 + lane%16]
// ---------------------------------------------------------------------------
__global__ void pack_weight(const float* __restrict__ W, _Float16* __restrict__ dst,
                            int K, int N) {
    int idx = blockIdx.x * blockDim.x + threadIdx.x;
    if (idx >= K * N) return;
    int j    = idx & 15;
    int lane = (idx >> 4) & 31;
    int tid  = idx >> 9;
    int ntn  = N >> 4;
    int kt = tid / ntn, nt = tid % ntn;
    int k = kt * 32 + ((j >> 3) << 4) + ((lane >> 4) << 3) + (j & 7);
    int n = nt * 16 + (lane & 15);
    dst[idx] = (_Float16)W[k * N + n];
}

// Convert h (fp32) to f16 once; every edge gather afterwards is conversion-free
// and half the bytes (2.5 MB buffer -> fully L2 resident).
__global__ void pack_h(const float* __restrict__ h, _Float16* __restrict__ h16, int n) {
    int idx = blockIdx.x * blockDim.x + threadIdx.x;
    if (idx < n) h16[idx] = (_Float16)h[idx];
}

__device__ __forceinline__ float silu(float x) {
    return x / (1.0f + __expf(-x));
}

// Load a 16x32 f16 A-fragment for this lane from an LDS row (row-major halves).
__device__ __forceinline__ v16h load_afrag(const _Float16* rowbase, int kt, int alo) {
    const _Float16* ap = rowbase + kt * 32 + alo;
    v16h a;
#pragma unroll
    for (int j = 0; j < 8; ++j) { a[j] = ap[j]; a[j + 8] = ap[16 + j]; }
    return a;
}

// ---------------------------------------------------------------------------
// Edge kernel: per wave, 16 edges.  e = silu(silu([h_row|h_col]@We1+b1)@We2+b2)
// then atomic scatter-add into agg[row].
// ---------------------------------------------------------------------------
__global__ __launch_bounds__(32 * WV)
void edge_kernel(const _Float16* __restrict__ h16,
                 const long long* __restrict__ eidx,
                 const _Float16* __restrict__ we1p, const float* __restrict__ be1,
                 const _Float16* __restrict__ we2p, const float* __restrict__ be2,
                 float* __restrict__ agg, int nEdges) {
    __shared__ alignas(16) _Float16 s_in [WV][16][2 * DD];  // 32 KB
    __shared__ alignas(16) _Float16 s_act[WV][16][DD];      // 16 KB

    const int lane = threadIdx.x & 31;
    const int w    = threadIdx.x >> 5;
    const int base = (blockIdx.x * WV + w) * 16;
    const long long* rowp = eidx;
    const long long* colp = eidx + nEdges;

    // ---- stage gathered f16 inputs into LDS: [16 edges][256] (pure 16B copies)
#pragma unroll
    for (int t = 0; t < 16; ++t) {
        int idx = t * 256 + lane * 8;          // over 16*256 halves, 8 per lane/iter
        int e   = idx >> 8;
        int k   = idx & 255;
        int eg  = base + e; if (eg >= nEdges) eg = 0;
        long long node = (k < DD) ? rowp[eg] : colp[eg];
        const float4 v = *(const float4*)(h16 + node * DD + (k & (DD - 1)));
        *(float4*)(&s_in[w][e][k]) = v;
    }
    __syncthreads();

    const int mrow = lane & 15;                 // A-fragment source row
    const int alo  = (lane & 16) ? 8 : 0;       // A-layout K offset for this lane half
    const int ncol = lane & 15;                 // C/D column
    const int mhi  = (lane & 16) ? 8 : 0;       // C/D row offset

    // ---- layer 1: [16x256] @ [256x128], K-tiles=8, N-tiles=8
#pragma unroll
    for (int nt = 0; nt < 8; ++nt) {
        v8f acc = {};                           // zero C -> inline-0 SRC2 form
#pragma unroll
        for (int kt = 0; kt < 8; ++kt) {
            v16h a = load_afrag(&s_in[w][mrow][0], kt, alo);
            v16h b = *(const v16h*)(we1p + (((kt << 3) + nt) << 9) + lane * 16);
            acc = __builtin_amdgcn_wmma_f32_16x16x32_f16(false, a, false, b,
                                                         (short)0, acc, false, false);
        }
        float bv = be1[nt * 16 + ncol];
#pragma unroll
        for (int r = 0; r < 8; ++r)
            s_act[w][r + mhi][nt * 16 + ncol] = (_Float16)silu(acc[r] + bv);
    }
    __syncthreads();

    int drow[8];
#pragma unroll
    for (int r = 0; r < 8; ++r) {
        int eg = base + r + mhi;
        drow[r] = (eg < nEdges) ? (int)rowp[eg] : -1;
    }

    // ---- layer 2: [16x128] @ [128x128], K-tiles=4, then silu + scatter-add
#pragma unroll
    for (int nt = 0; nt < 8; ++nt) {
        v8f acc = {};
#pragma unroll
        for (int kt = 0; kt < 4; ++kt) {
            v16h a = load_afrag(&s_act[w][mrow][0], kt, alo);
            v16h b = *(const v16h*)(we2p + (((kt << 3) + nt) << 9) + lane * 16);
            acc = __builtin_amdgcn_wmma_f32_16x16x32_f16(false, a, false, b,
                                                         (short)0, acc, false, false);
        }
        float bv = be2[nt * 16 + ncol];
#pragma unroll
        for (int r = 0; r < 8; ++r) {
            if (drow[r] >= 0)
                atomicAdd(&agg[drow[r] * DD + nt * 16 + ncol], silu(acc[r] + bv));
        }
    }
}

// ---------------------------------------------------------------------------
// Node kernel: per wave, 16 nodes. out = h + silu([h|agg/100]@Wn1+b1)@Wn2+b2
// ---------------------------------------------------------------------------
__global__ __launch_bounds__(32 * WV)
void node_kernel(const float* __restrict__ h, const _Float16* __restrict__ h16,
                 const float* __restrict__ agg,
                 const _Float16* __restrict__ wn1p, const float* __restrict__ bn1,
                 const _Float16* __restrict__ wn2p, const float* __restrict__ bn2,
                 float* __restrict__ out, int nNodes) {
    __shared__ alignas(16) _Float16 s_in [WV][16][2 * DD];
    __shared__ alignas(16) _Float16 s_act[WV][16][DD];

    const int lane = threadIdx.x & 31;
    const int w    = threadIdx.x >> 5;
    const int base = (blockIdx.x * WV + w) * 16;

    // ---- stage [h16 | agg/100] into LDS
#pragma unroll
    for (int t = 0; t < 16; ++t) {
        int idx = t * 256 + lane * 8;
        int i   = idx >> 8;
        int k   = idx & 255;
        int node = base + i; if (node >= nNodes) node = 0;
        if (k < DD) {
            const float4 v = *(const float4*)(h16 + node * DD + k);
            *(float4*)(&s_in[w][i][k]) = v;
        } else {
            const float* src = agg + node * DD + (k - DD);
            const float4 v0 = *(const float4*)(src);
            const float4 v1 = *(const float4*)(src + 4);
            _Float16* d = &s_in[w][i][k];
            d[0] = (_Float16)(v0.x * 0.01f); d[1] = (_Float16)(v0.y * 0.01f);
            d[2] = (_Float16)(v0.z * 0.01f); d[3] = (_Float16)(v0.w * 0.01f);
            d[4] = (_Float16)(v1.x * 0.01f); d[5] = (_Float16)(v1.y * 0.01f);
            d[6] = (_Float16)(v1.z * 0.01f); d[7] = (_Float16)(v1.w * 0.01f);
        }
    }
    __syncthreads();

    const int mrow = lane & 15;
    const int alo  = (lane & 16) ? 8 : 0;
    const int ncol = lane & 15;
    const int mhi  = (lane & 16) ? 8 : 0;

    // ---- layer 1 with SiLU
#pragma unroll
    for (int nt = 0; nt < 8; ++nt) {
        v8f acc = {};
#pragma unroll
        for (int kt = 0; kt < 8; ++kt) {
            v16h a = load_afrag(&s_in[w][mrow][0], kt, alo);
            v16h b = *(const v16h*)(wn1p + (((kt << 3) + nt) << 9) + lane * 16);
            acc = __builtin_amdgcn_wmma_f32_16x16x32_f16(false, a, false, b,
                                                         (short)0, acc, false, false);
        }
        float bv = bn1[nt * 16 + ncol];
#pragma unroll
        for (int r = 0; r < 8; ++r)
            s_act[w][r + mhi][nt * 16 + ncol] = (_Float16)silu(acc[r] + bv);
    }
    __syncthreads();

    // ---- layer 2 (no activation) + residual store
#pragma unroll
    for (int nt = 0; nt < 8; ++nt) {
        v8f acc = {};
#pragma unroll
        for (int kt = 0; kt < 4; ++kt) {
            v16h a = load_afrag(&s_act[w][mrow][0], kt, alo);
            v16h b = *(const v16h*)(wn2p + (((kt << 3) + nt) << 9) + lane * 16);
            acc = __builtin_amdgcn_wmma_f32_16x16x32_f16(false, a, false, b,
                                                         (short)0, acc, false, false);
        }
        float bv = bn2[nt * 16 + ncol];
#pragma unroll
        for (int r = 0; r < 8; ++r) {
            int node = base + r + mhi;
            if (node < nNodes) {
                int o = node * DD + nt * 16 + ncol;
                out[o] = h[o] + acc[r] + bv;
            }
        }
    }
}

// ---------------------------------------------------------------------------
extern "C" void kernel_launch(void* const* d_in, const int* in_sizes, int n_in,
                              void* d_out, int out_size, void* d_ws, size_t ws_size,
                              hipStream_t stream) {
    const float*     h    = (const float*)d_in[0];
    const long long* eidx = (const long long*)d_in[1];  // int64 [2, E]
    const float* we1 = (const float*)d_in[2];
    const float* be1 = (const float*)d_in[3];
    const float* we2 = (const float*)d_in[4];
    const float* be2 = (const float*)d_in[5];
    const float* wn1 = (const float*)d_in[6];
    const float* bn1 = (const float*)d_in[7];
    const float* wn2 = (const float*)d_in[8];
    const float* bn2 = (const float*)d_in[9];
    float* out = (float*)d_out;

    const int nNodes = in_sizes[0] / DD;   // 10000
    const int nEdges = in_sizes[1] / 2;    // 640000

    char* ws = (char*)d_ws;
    float* agg = (float*)ws;
    size_t aggBytes = (size_t)nNodes * DD * sizeof(float);
    size_t o = (aggBytes + 255) & ~(size_t)255;
    _Float16* h16  = (_Float16*)(ws + o); o += (size_t)nNodes * DD * 2;
    o = (o + 255) & ~(size_t)255;
    _Float16* we1p = (_Float16*)(ws + o); o += (size_t)2 * DD * DD * 2;
    _Float16* we2p = (_Float16*)(ws + o); o += (size_t)DD * DD * 2;
    _Float16* wn1p = (_Float16*)(ws + o); o += (size_t)2 * DD * DD * 2;
    _Float16* wn2p = (_Float16*)(ws + o); o += (size_t)DD * DD * 2;
    (void)ws_size; (void)n_in; (void)out_size;

    hipMemsetAsync(agg, 0, aggBytes, stream);

    pack_h<<<(nNodes * DD + 255) / 256, 256, 0, stream>>>(h, h16, nNodes * DD);
    pack_weight<<<(2 * DD * DD + 255) / 256, 256, 0, stream>>>(we1, we1p, 2 * DD, DD);
    pack_weight<<<(DD * DD + 255) / 256, 256, 0, stream>>>(we2, we2p, DD, DD);
    pack_weight<<<(2 * DD * DD + 255) / 256, 256, 0, stream>>>(wn1, wn1p, 2 * DD, DD);
    pack_weight<<<(DD * DD + 255) / 256, 256, 0, stream>>>(wn2, wn2p, DD, DD);

    int eBlocks = (nEdges + 16 * WV - 1) / (16 * WV);   // 10000
    edge_kernel<<<eBlocks, 32 * WV, 0, stream>>>(h16, eidx, we1p, be1, we2p, be2,
                                                 agg, nEdges);

    int nBlocks = (nNodes + 16 * WV - 1) / (16 * WV);   // 157
    node_kernel<<<nBlocks, 32 * WV, 0, stream>>>(h, h16, agg, wn1p, bn1, wn2p, bn2,
                                                 out, nNodes);
}
